// PlayerGraphSAGE_46583215292451
// MI455X (gfx1250) — compile-verified, matrix operand order checked by reading, not run on hardware
//
#include <hip/hip_runtime.h>
#include <hip/hip_bf16.h>

// ---------------------------------------------------------------------------
// GraphSAGE (3 layers, mean aggregation) for MI455X / gfx1250.
//   agg = segment_mean(x[src], dst);  h = relu(agg@Wl^T + bl + x@Wr^T)
// Scatter/gather is the roofline bottleneck (L2-resident f32 atomics).
// GEMMs run on the bf16 WMMA pipe (V_WMMA_F32_16X16X32_BF16) with a bf16x3
// split (a ~ a_hi + a_lo) for ~fp32 accuracy. Weights are pre-packed into
// WMMA B-fragment lane layout (hi/lo bf16) ONCE per launch, so the GEMM
// inner loop is wide b128 loads + WMMAs with minimal VALU.
// ---------------------------------------------------------------------------

typedef __bf16 bf16_t;
typedef bf16_t v16bf __attribute__((ext_vector_type(16)));
typedef float  v8f   __attribute__((ext_vector_type(8)));

// Build one 16x32 bf16 WMMA fragment slice for this lane from a row-major f32
// matrix, split into hi+lo bf16 (bf16x3 trick).
// ISA 7.12.2 (16-bit A 16x32): lane&15 = M (N for B), half = lane>>4 selects
// K-half: slots 0..7 hold K = half*8 + {0..7}, slots 8..15 hold the same +16.
__device__ __forceinline__ void load_frag_bf16x2(const float* __restrict__ rowPtr,
                                                 float scale, int half,
                                                 v16bf& hi, v16bf& lo)
{
  const float4* p = (const float4*)(rowPtr + half * 8);
  float4 f[4];
  f[0] = p[0];  f[1] = p[1];   // K = half*8 + 0..7   -> slots 0..7
  f[2] = p[4];  f[3] = p[5];   // K = half*8 + 16..23 -> slots 8..15
  const float* fv = (const float*)f;
#pragma unroll
  for (int i = 0; i < 16; ++i) {
    const float a = fv[i] * scale;
    const bf16_t h = (bf16_t)a;
    hi[i] = h;
    lo[i] = (bf16_t)(a - (float)h);
  }
}

#define WMMA_BF16(A, B, C) \
  __builtin_amdgcn_wmma_f32_16x16x32_bf16(false, (A), false, (B), (short)0, (C), false, false)

// Pre-pack a weight matrix W[dout,K] (row-major f32) into WMMA B-fragment
// layout: fragment fi = ct*(K/32)+kb holds 32 lanes x (16 hi + 16 lo) bf16,
// 64 B per lane, fragment-major so the GEMM k-loop streams contiguously.
// One wave per fragment.
__launch_bounds__(32)
__global__ void pack_w_kernel(const float* __restrict__ W,
                              bf16_t* __restrict__ Wp, int K)
{
  const int lane = threadIdx.x;
  const int r    = lane & 15;
  const int half = lane >> 4;
  const int ct   = blockIdx.x;          // 16-row tile of W (= output col tile)
  const int kb   = blockIdx.y;          // 32-wide K tile
  const int nkb  = gridDim.y;

  v16bf hi, lo;
  load_frag_bf16x2(W + (size_t)(ct * 16 + r) * K + kb * 32, 1.0f, half, hi, lo);

  bf16_t* base = Wp + ((size_t)(ct * nkb + kb) * 32 + lane) * 32;
  *(v16bf*)(base)      = hi;
  *(v16bf*)(base + 16) = lo;
}

// One wave computes a 16x64 tile of:
//   Out = (Agg * cntInv[row]) @ Wl^T + bl + X @ Wr^T   (+ optional ReLU)
// Wlp/Wrp are pre-packed bf16 hi/lo fragments.
__launch_bounds__(32)
__global__ void sage_gemm_wmma(const float* __restrict__ Agg,
                               const float* __restrict__ cntInv,
                               const float* __restrict__ X,
                               const bf16_t* __restrict__ Wlp,
                               const float* __restrict__ bl,
                               const bf16_t* __restrict__ Wrp,
                               float* __restrict__ Out,
                               int K, int dout, int doRelu)
{
  const int lane = threadIdx.x;
  const int r    = lane & 15;
  const int half = lane >> 4;

  const int rowBase = blockIdx.x * 16;
  const int colBase = blockIdx.y * 64;
  const int nkb     = K >> 5;

  v8f acc[4] = {v8f{}, v8f{}, v8f{}, v8f{}};

  const float invDeg = cntInv[rowBase + r];

#pragma unroll
  for (int pass = 0; pass < 2; ++pass) {
    const float*  Am = pass ? X   : Agg;
    const bf16_t* Wp = pass ? Wrp : Wlp;
    const float   sc = pass ? 1.0f : invDeg;
    const float* aRow = Am + (size_t)(rowBase + r) * K;

    for (int kb = 0; kb < nkb; ++kb) {
      v16bf ahi, alo;
      load_frag_bf16x2(aRow + kb * 32, sc, half, ahi, alo);

#pragma unroll
      for (int t = 0; t < 4; ++t) {
        const int ct = (colBase >> 4) + t;
        const bf16_t* bp = Wp + ((size_t)(ct * nkb + kb) * 32 + lane) * 32;
        const v16bf bhi = *(const v16bf*)(bp);
        const v16bf blo = *(const v16bf*)(bp + 16);
        acc[t] = WMMA_BF16(ahi, bhi, acc[t]);
        acc[t] = WMMA_BF16(ahi, blo, acc[t]);
        acc[t] = WMMA_BF16(alo, bhi, acc[t]);
      }
    }
  }

  // Epilogue: C/D layout (ISA 7.12.2): lane&15 = N, VGPR vr = M (lanes>=16: +8)
  const int rowOut = rowBase + half * 8;
#pragma unroll
  for (int t = 0; t < 4; ++t) {
    const int col = colBase + t * 16 + r;
    const float b = bl[col];
#pragma unroll
    for (int vr = 0; vr < 8; ++vr) {
      float v = acc[t][vr] + b;
      if (doRelu) v = fmaxf(v, 0.0f);
      Out[(size_t)(rowOut + vr) * dout + col] = v;
    }
  }
}

// Edge scatter: one thread per (edge, feature). Consecutive lanes handle
// consecutive features of one edge -> coalesced gather from x[src] and
// coalesced global_atomic_add_f32 into agg[dst] (L2-resident RMW).
__global__ void scatter_add_kernel(const float* __restrict__ Xin,
                                   const long long* __restrict__ src,
                                   const long long* __restrict__ dst,
                                   float* __restrict__ Agg,
                                   long long total, int fshift, int fmask)
{
  const long long idx = (long long)blockIdx.x * blockDim.x + threadIdx.x;
  if (idx >= total) return;
  const long long e = idx >> fshift;
  const int       f = (int)(idx & fmask);
  const int       F = fmask + 1;
  const long long s = src[e];
  const long long d = dst[e];
  unsafeAtomicAdd(&Agg[(size_t)d * F + f], Xin[(size_t)s * F + f]);
}

__global__ void degree_kernel(const long long* __restrict__ dst,
                              float* __restrict__ cnt, int E)
{
  const int e = blockIdx.x * blockDim.x + threadIdx.x;
  if (e < E) unsafeAtomicAdd(&cnt[dst[e]], 1.0f);
}

__global__ void recip_kernel(float* __restrict__ cnt, int n)
{
  const int i = blockIdx.x * blockDim.x + threadIdx.x;
  if (i < n) cnt[i] = 1.0f / fmaxf(cnt[i], 1.0f);
}

// ---------------------------------------------------------------------------

extern "C" void kernel_launch(void* const* d_in, const int* in_sizes, int n_in,
                              void* d_out, int out_size, void* d_ws, size_t ws_size,
                              hipStream_t stream)
{
  const float*     x   = (const float*)d_in[0];
  const long long* ei  = (const long long*)d_in[1];
  const float*     Wl0 = (const float*)d_in[2];
  const float*     bl0 = (const float*)d_in[3];
  const float*     Wr0 = (const float*)d_in[4];
  const float*     Wl1 = (const float*)d_in[5];
  const float*     bl1 = (const float*)d_in[6];
  const float*     Wr1 = (const float*)d_in[7];
  const float*     Wl2 = (const float*)d_in[8];
  const float*     bl2 = (const float*)d_in[9];
  const float*     Wr2 = (const float*)d_in[10];
  float*           out = (float*)d_out;

  const int N = in_sizes[0] / 128;   // 50000 nodes
  const int E = in_sizes[1] / 2;     // 800000 edges
  const long long* src = ei;
  const long long* dst = ei + E;

  // Workspace: cntInv | agg(f32) | h0 | h1 | packed weights (hi+lo bf16 = 4B/elt)
  char*  ws = (char*)d_ws;
  size_t o  = 0;
  auto carve = [&](size_t bytes) -> char* {
    char* p = ws + o;
    o += (bytes + 255) & ~(size_t)255;
    return p;
  };
  float*  cntInv = (float*)carve((size_t)N * 4);
  float*  agg    = (float*)carve((size_t)N * 512 * 4);
  float*  h0     = (float*)carve((size_t)N * 512 * 4);
  float*  h1     = (float*)carve((size_t)N * 512 * 4);
  bf16_t* Wlp0   = (bf16_t*)carve((size_t)512 * 128 * 4);
  bf16_t* Wrp0   = (bf16_t*)carve((size_t)512 * 128 * 4);
  bf16_t* Wlp1   = (bf16_t*)carve((size_t)512 * 512 * 4);
  bf16_t* Wrp1   = (bf16_t*)carve((size_t)512 * 512 * 4);
  bf16_t* Wlp2   = (bf16_t*)carve((size_t)256 * 512 * 4);
  bf16_t* Wrp2   = (bf16_t*)carve((size_t)256 * 512 * 4);
  (void)ws_size; (void)n_in; (void)out_size;

  // --- pre-pack all weights into WMMA fragment layout (one-shot, ~4 MB) ---
  pack_w_kernel<<<dim3(512 / 16, 128 / 32), 32, 0, stream>>>(Wl0, Wlp0, 128);
  pack_w_kernel<<<dim3(512 / 16, 128 / 32), 32, 0, stream>>>(Wr0, Wrp0, 128);
  pack_w_kernel<<<dim3(512 / 16, 512 / 32), 32, 0, stream>>>(Wl1, Wlp1, 512);
  pack_w_kernel<<<dim3(512 / 16, 512 / 32), 32, 0, stream>>>(Wr1, Wrp1, 512);
  pack_w_kernel<<<dim3(256 / 16, 512 / 32), 32, 0, stream>>>(Wl2, Wlp2, 512);
  pack_w_kernel<<<dim3(256 / 16, 512 / 32), 32, 0, stream>>>(Wr2, Wrp2, 512);

  // --- mean-aggregation normalizers (1 / max(deg,1)) ---
  hipMemsetAsync(cntInv, 0, (size_t)N * 4, stream);
  degree_kernel<<<(E + 255) / 256, 256, 0, stream>>>(dst, cntInv, E);
  recip_kernel<<<(N + 255) / 256, 256, 0, stream>>>(cntInv, N);

  // --- layer 0: K=128 -> 512, ReLU ---
  {
    const int F = 128;
    hipMemsetAsync(agg, 0, (size_t)N * F * 4, stream);
    const long long total = (long long)E * F;
    scatter_add_kernel<<<(unsigned)((total + 255) / 256), 256, 0, stream>>>(
        x, src, dst, agg, total, 7, F - 1);
    dim3 g(N / 16, 512 / 64);
    sage_gemm_wmma<<<g, 32, 0, stream>>>(agg, cntInv, x, Wlp0, bl0, Wrp0, h0,
                                         128, 512, 1);
  }

  // --- layer 1: K=512 -> 512, ReLU ---
  {
    const int F = 512;
    hipMemsetAsync(agg, 0, (size_t)N * F * 4, stream);
    const long long total = (long long)E * F;
    scatter_add_kernel<<<(unsigned)((total + 255) / 256), 256, 0, stream>>>(
        h0, src, dst, agg, total, 9, F - 1);
    dim3 g(N / 16, 512 / 64);
    sage_gemm_wmma<<<g, 32, 0, stream>>>(agg, cntInv, h0, Wlp1, bl1, Wrp1, h1,
                                         512, 512, 1);
  }

  // --- layer 2: K=512 -> 256, no activation ---
  {
    const int F = 512;
    hipMemsetAsync(agg, 0, (size_t)N * F * 4, stream);
    const long long total = (long long)E * F;
    scatter_add_kernel<<<(unsigned)((total + 255) / 256), 256, 0, stream>>>(
        h1, src, dst, agg, total, 9, F - 1);
    dim3 g(N / 16, 256 / 64);
    sage_gemm_wmma<<<g, 32, 0, stream>>>(agg, cntInv, h1, Wlp2, bl2, Wrp2, out,
                                         512, 256, 0);
  }
}
